// BahdanauAttention_15247133901618
// MI455X (gfx1250) — matrix operand order repaired and checked by reading, not compiled, not run on hardware
//
#include <hip/hip_runtime.h>
#include <hip/hip_bf16.h>

#define B_    64
#define S_    1024
#define HID_  1024
#define ENC_  2048
#define NEGV  (-1e10f)

typedef __attribute__((ext_vector_type(16))) __bf16 v16bf;
typedef __attribute__((ext_vector_type(8)))  float  v8f;

// ---------------------------------------------------------------------------
// Kernel 1: Wk fp32 [HID][ENC] -> bf16 [HID][ENC] (row-major, K contiguous).
// ---------------------------------------------------------------------------
__global__ void cvt_wk_kernel(const float* __restrict__ wk,
                              unsigned short* __restrict__ wkb_raw) {
    __bf16* wkb = (__bf16*)wkb_raw;
    int i = (blockIdx.x * 256 + threadIdx.x) * 4;
    float4 f = *(const float4*)(wk + i);
    wkb[i + 0] = (__bf16)f.x;
    wkb[i + 1] = (__bf16)f.y;
    wkb[i + 2] = (__bf16)f.z;
    wkb[i + 3] = (__bf16)f.w;
}

// ---------------------------------------------------------------------------
// Kernel 2: q[b,h] = sum_i dec[b,i] * Wq[h,i]   (tiny fp32 GEMM, 134 MFLOP)
// ---------------------------------------------------------------------------
__global__ void query_kernel(const float* __restrict__ dec,
                             const float* __restrict__ wq,
                             float* __restrict__ q) {
    __shared__ float ds[HID_];
    int b = blockIdx.y;
    int h = blockIdx.x * 256 + threadIdx.x;
    for (int i = threadIdx.x; i < HID_; i += 256) ds[i] = dec[(size_t)b * HID_ + i];
    __syncthreads();
    const float* wr = wq + (size_t)h * HID_;
    float acc = 0.f;
    for (int i = 0; i < HID_; i += 4) {
        float4 w4 = *(const float4*)(wr + i);
        acc += w4.x * ds[i] + w4.y * ds[i + 1] + w4.z * ds[i + 2] + w4.w * ds[i + 3];
    }
    q[(size_t)b * HID_ + h] = acc;
}

// ---------------------------------------------------------------------------
// Kernel 3: fused  energy[b,s] = sum_h tanh(q[b,h] + (enc[b] x Wk^T)[s,h]) * v[h]
//
// WG = 256 threads = 8 waves. WG covers 32 rows of S (2 M-tiles, shared by all
// waves); wave w covers N columns [w*128, w*128+128) (8 N-tiles). Together the
// 8 waves span all HID=1024 columns, so the tanh-dot reduction completes with
// one LDS reduction (no global atomics) and `keys` is never written to memory.
//
// WMMA bf16 16x16x32, A = enc tile (M=s,K=e) converted fp32->bf16 in regs,
// B = Wk_bf (K=e contiguous per row n) loaded as one 32-byte chunk per lane.
// ---------------------------------------------------------------------------
__global__ void __launch_bounds__(256)
energy_kernel(const float* __restrict__ enc,
              const unsigned short* __restrict__ wkb_raw,
              const float* __restrict__ q,
              const float* __restrict__ vvec,
              float* __restrict__ energy) {
    const __bf16* wkb = (const __bf16*)wkb_raw;

    const int b      = blockIdx.y;
    const int row0   = blockIdx.x * 32;        // first S row of this WG
    const int wave   = threadIdx.x >> 5;       // 0..7  -> N chunk
    const int lane   = threadIdx.x & 31;
    const int lane15 = lane & 15;
    const int laneHi = lane >> 4;              // 0 or 1

    v8f acc[2][8] = {};                        // 2 M-tiles x 8 N-tiles

    // A: rows row0 + mt*16 + lane15 (lanes 0-15) / same rows for lanes 16-31,
    //    K runs at k0 + laneHi*16, 16 contiguous (contiguous k order, same as B).
    const float* abase = enc + ((size_t)b * S_ + row0 + lane15) * ENC_;
    // B: row n = wave*128 + nt*16 + lane15, K offset k0 + laneHi*16,
    //    16 contiguous bf16 = 32 bytes per lane (contiguous k order, same as A).
    const __bf16* bbase = wkb + (size_t)(wave * 128 + lane15) * ENC_ + laneHi * 16;

    for (int k0 = 0; k0 < ENC_; k0 += 32) {
        v16bf afr[2];
#pragma unroll
        for (int mt = 0; mt < 2; ++mt) {
            const float* ap = abase + (size_t)mt * 16 * ENC_ + k0 + laneHi * 16;
            float4 f0 = *(const float4*)(ap);
            float4 f1 = *(const float4*)(ap + 4);
            float4 f2 = *(const float4*)(ap + 8);
            float4 f3 = *(const float4*)(ap + 12);
            v16bf a;
            a[0]  = (__bf16)f0.x; a[1]  = (__bf16)f0.y; a[2]  = (__bf16)f0.z; a[3]  = (__bf16)f0.w;
            a[4]  = (__bf16)f1.x; a[5]  = (__bf16)f1.y; a[6]  = (__bf16)f1.z; a[7]  = (__bf16)f1.w;
            a[8]  = (__bf16)f2.x; a[9]  = (__bf16)f2.y; a[10] = (__bf16)f2.z; a[11] = (__bf16)f2.w;
            a[12] = (__bf16)f3.x; a[13] = (__bf16)f3.y; a[14] = (__bf16)f3.z; a[15] = (__bf16)f3.w;
            afr[mt] = a;
        }
#pragma unroll
        for (int nt = 0; nt < 8; ++nt) {
            v16bf bfr = *(const v16bf*)(bbase + (size_t)nt * 16 * ENC_ + k0);
#pragma unroll
            for (int mt = 0; mt < 2; ++mt) {
                acc[mt][nt] = __builtin_amdgcn_wmma_f32_16x16x32_bf16(
                    false, afr[mt], false, bfr, (short)0, acc[mt][nt],
                    false, false);
            }
        }
    }

    // Epilogue: energy contribution. C layout: VGPR j -> row M = j + laneHi*8,
    // column N = lane15 within the 16x16 tile.
    float qn[8], vn[8];
#pragma unroll
    for (int nt = 0; nt < 8; ++nt) {
        int n  = wave * 128 + nt * 16 + lane15;
        qn[nt] = q[(size_t)b * HID_ + n];
        vn[nt] = vvec[n];
    }

    __shared__ float esm[32];
    if (threadIdx.x < 32) esm[threadIdx.x] = 0.f;
    __syncthreads();

#pragma unroll
    for (int mt = 0; mt < 2; ++mt) {
#pragma unroll
        for (int j = 0; j < 8; ++j) {
            float part = 0.f;
#pragma unroll
            for (int nt = 0; nt < 8; ++nt)
                part += tanhf(qn[nt] + acc[mt][nt][j]) * vn[nt];
            int row = mt * 16 + j + laneHi * 8;
            atomicAdd(&esm[row], part);              // ds_add_f32
        }
    }
    __syncthreads();

    if (threadIdx.x < 32)
        energy[(size_t)b * S_ + row0 + threadIdx.x] = esm[threadIdx.x];
}

// ---------------------------------------------------------------------------
// Kernel 4: masked softmax over S per batch (one WG per b).
// ---------------------------------------------------------------------------
__global__ void softmax_kernel(const float* __restrict__ energy,
                               const int* __restrict__ mask,
                               float* __restrict__ attn) {
    __shared__ float red[256];
    int b = blockIdx.x;
    float e[4];
    float mx = -3.4e38f;
#pragma unroll
    for (int i = 0; i < 4; ++i) {
        int s = threadIdx.x + i * 256;
        float ev = energy[(size_t)b * S_ + s];
        if (mask[(size_t)b * S_ + s] == 0) ev = NEGV;
        e[i] = ev;
        mx = fmaxf(mx, ev);
    }
    red[threadIdx.x] = mx;
    __syncthreads();
    for (int off = 128; off > 0; off >>= 1) {
        if (threadIdx.x < off)
            red[threadIdx.x] = fmaxf(red[threadIdx.x], red[threadIdx.x + off]);
        __syncthreads();
    }
    mx = red[0];
    __syncthreads();

    float sum = 0.f;
#pragma unroll
    for (int i = 0; i < 4; ++i) { e[i] = __expf(e[i] - mx); sum += e[i]; }
    red[threadIdx.x] = sum;
    __syncthreads();
    for (int off = 128; off > 0; off >>= 1) {
        if (threadIdx.x < off)
            red[threadIdx.x] = red[threadIdx.x] + red[threadIdx.x + off];
        __syncthreads();
    }
    float inv = 1.f / red[0];
#pragma unroll
    for (int i = 0; i < 4; ++i)
        attn[(size_t)b * S_ + threadIdx.x + i * 256] = e[i] * inv;
}

// ---------------------------------------------------------------------------
// Kernel 5: context[b,e] = sum_s attn[b,s] * enc[b,s,e].  Pure bandwidth:
// second 512 MB streaming read of enc, fully coalesced, attn held in LDS.
// ---------------------------------------------------------------------------
__global__ void context_kernel(const float* __restrict__ enc,
                               const float* __restrict__ attn,
                               float* __restrict__ ctx) {
    __shared__ float ws[S_];
    int b = blockIdx.y;
    int e = blockIdx.x * 256 + threadIdx.x;
    for (int s = threadIdx.x; s < S_; s += 256) ws[s] = attn[(size_t)b * S_ + s];
    __syncthreads();
    float acc = 0.f;
    const float* ep = enc + (size_t)b * S_ * ENC_ + e;
    for (int s = 0; s < S_; ++s) acc += ws[s] * ep[(size_t)s * ENC_];
    ctx[(size_t)b * ENC_ + e] = acc;
}

// ---------------------------------------------------------------------------
extern "C" void kernel_launch(void* const* d_in, const int* in_sizes, int n_in,
                              void* d_out, int out_size, void* d_ws, size_t ws_size,
                              hipStream_t stream) {
    (void)in_sizes; (void)n_in; (void)out_size; (void)ws_size;

    const float* dec  = (const float*)d_in[0];   // [B, HID]
    const float* enc  = (const float*)d_in[1];   // [B, S, ENC]
    const int*   mask = (const int*)d_in[2];     // [B, S]
    const float* wq   = (const float*)d_in[3];   // [HID, HID]
    const float* wk   = (const float*)d_in[4];   // [HID, ENC]
    const float* vvec = (const float*)d_in[5];   // [HID]

    float* out  = (float*)d_out;
    float* ctx  = out;                   // [B, ENC]  (first tuple element)
    float* attn = out + (size_t)B_ * ENC_; // [B, S]  (second tuple element)

    char* ws = (char*)d_ws;
    unsigned short* wkb   = (unsigned short*)ws;                         // 4 MB bf16 Wk
    float*          q     = (float*)(ws + (size_t)HID_ * ENC_ * 2);      // 256 KB
    float*          energ = (float*)(ws + (size_t)HID_ * ENC_ * 2
                                        + (size_t)B_ * HID_ * 4);        // 256 KB

    cvt_wk_kernel<<<(HID_ * ENC_) / (256 * 4), 256, 0, stream>>>(wk, wkb);
    query_kernel<<<dim3(HID_ / 256, B_), 256, 0, stream>>>(dec, wq, q);
    energy_kernel<<<dim3(S_ / 32, B_), 256, 0, stream>>>(enc, wkb, q, vvec, energ);
    softmax_kernel<<<B_, 256, 0, stream>>>(energ, mask, attn);
    context_kernel<<<dim3(ENC_ / 256, B_), 256, 0, stream>>>(enc, attn, ctx);
}
